// Attention_85899346085
// MI455X (gfx1250) — compile-verified, matrix operand order checked by reading, not run on hardware
//
#include <hip/hip_runtime.h>
#include <hip/hip_bf16.h>

// GPT-2 attention block for MI455X (gfx1250), bf16 WMMA path + async LDS staging.
// Inputs (fp32): x(2,2048,1024), c_attn_w(1024,3072), c_attn_b(3072),
//                c_proj_w(1024,1024), c_proj_b(1024)
// Output (fp32, concatenated): a(2,2048,1024) [4,194,304 floats],
//                              present(2,2,16,2048,64) [8,388,608 floats]
// Workspace (bf16): Q(B,H,S,hd) | K(B,H,S,hd) | V^T(B,H,hd,S) | attn_out(B,S,D)
//   = 4 * 4,194,304 bf16 = 32 MB

typedef __bf16 bf16;
typedef __attribute__((ext_vector_type(8)))  __bf16 v8bf;
typedef __attribute__((ext_vector_type(16))) __bf16 v16bf;
typedef __attribute__((ext_vector_type(8)))  float  v8f;

#define S_LEN 2048
#define D_MODEL 1024
#define N3 3072
#define N_HEAD 16
#define HEAD_D 64

// fp32 -> bf16 round-to-nearest-even
__device__ __forceinline__ bf16 f2bf(float x) {
  unsigned u = __builtin_bit_cast(unsigned, x);
  unsigned r = u + 0x7FFFu + ((u >> 16) & 1u);
  unsigned short h = (unsigned short)(r >> 16);
  return __builtin_bit_cast(bf16, h);
}

__device__ __forceinline__ unsigned pk2(float a, float b) {
  unsigned short ha = __builtin_bit_cast(unsigned short, f2bf(a));
  unsigned short hb = __builtin_bit_cast(unsigned short, f2bf(b));
  return (unsigned)ha | ((unsigned)hb << 16);
}

__device__ __forceinline__ v8f wmma_bf16(v16bf a, v16bf b, v8f c) {
  // (neg_a, A, neg_b, B, c_mod, C, reuse_a, reuse_b)
  return __builtin_amdgcn_wmma_f32_16x16x32_bf16(false, a, false, b, (short)0, c,
                                                 false, false);
}

// CDNA5 async global->LDS 16B copy (GVS mode), tracked by ASYNCcnt.
// lds_addr: low 32 bits of the generic shared pointer (ISA 10.2: LDS aperture
// addresses truncate to the 32-bit LDS byte offset). voff: 32-bit byte offset
// added to the uniform 64-bit SGPR base.
__device__ __forceinline__ void async_b128(const void* base, unsigned voff,
                                           void* lds_ptr) {
  unsigned lds_addr = (unsigned)(size_t)lds_ptr;
  asm volatile("global_load_async_to_lds_b128 %0, %1, %2"
               :
               : "v"(lds_addr), "v"(voff), "s"(base)
               : "memory");
}

__device__ __forceinline__ void wait_async0() {
  asm volatile("s_wait_asynccnt 0x0" ::: "memory");
}

// Load a 16x32 A-fragment (or the mirrored B-fragment: lane n reads row n of the
// [N][K]-stored matrix) per ISA 7.12.2: lane's row = row0 + (lane&15); K chunks at
// koff + half*8 and koff + 16 + half*8, each 8 contiguous bf16 (one b128 load).
__device__ __forceinline__ v16bf load_frag(const bf16* base, int row0, int ld,
                                           int koff, int lane) {
  const int row = row0 + (lane & 15);
  const int off = row * ld + koff + ((lane >> 4) << 3);
  v8bf lo = *(const v8bf*)(base + off);
  v8bf hi = *(const v8bf*)(base + off + 16);
  v16bf r;
#pragma unroll
  for (int i = 0; i < 8; ++i) { r[i] = lo[i]; r[8 + i] = hi[i]; }
  return r;
}

__device__ __forceinline__ v8f zero8() {
  v8f z;
#pragma unroll
  for (int e = 0; e < 8; ++e) z[e] = 0.0f;
  return z;
}

// ---------------------------------------------------------------------------
// Kernel 1: qkv = x @ c_attn_w + b, scatter to Q/K/V^T (bf16) + present (fp32)
// Block tile 128x128, K-tile 64. 8 waves in 2x4; wave tile 64x32 = 4x2 WMMA.
// ---------------------------------------------------------------------------
__global__ __launch_bounds__(256) void qkv_gemm(
    const float* __restrict__ x, const float* __restrict__ w,
    const float* __restrict__ bias, bf16* __restrict__ qws,
    bf16* __restrict__ kws, bf16* __restrict__ vtws,
    float* __restrict__ presK, float* __restrict__ presV) {
  __shared__ __align__(16) bf16 As[128 * 72];   // [m][k]
  __shared__ __align__(16) bf16 Bs[128 * 72];   // [n][k]  (W transposed)
  const int tid = threadIdx.x, lane = tid & 31, wave = tid >> 5;
  const int n0 = blockIdx.x * 128, m0 = blockIdx.y * 128;
  const int mw = (wave >> 2) * 64, nw = (wave & 3) * 32;

  v8f acc[4][2];
#pragma unroll
  for (int i = 0; i < 4; ++i)
#pragma unroll
    for (int j = 0; j < 2; ++j) acc[i][j] = zero8();

  for (int kc = 0; kc < D_MODEL; kc += 64) {
    // x tile 128x64 (fp32 -> bf16), coalesced float4 loads
#pragma unroll
    for (int p = 0; p < 8; ++p) {
      int slot = tid + p * 256;
      int r = slot >> 4, c = (slot & 15) << 2;
      float4 v = *(const float4*)(x + (long)(m0 + r) * D_MODEL + kc + c);
      *(uint2*)(As + r * 72 + c) = make_uint2(pk2(v.x, v.y), pk2(v.z, v.w));
    }
    // W tile 64x128 -> Bs[n][k] transposed
#pragma unroll
    for (int p = 0; p < 8; ++p) {
      int slot = tid + p * 256;
      int k = slot >> 5, n = (slot & 31) << 2;
      float4 v = *(const float4*)(w + (long)(kc + k) * N3 + n0 + n);
      Bs[(n + 0) * 72 + k] = f2bf(v.x);
      Bs[(n + 1) * 72 + k] = f2bf(v.y);
      Bs[(n + 2) * 72 + k] = f2bf(v.z);
      Bs[(n + 3) * 72 + k] = f2bf(v.w);
    }
    __syncthreads();
#pragma unroll
    for (int kk = 0; kk < 64; kk += 32) {
      v16bf bf0 = load_frag(Bs, nw + 0, 72, kk, lane);
      v16bf bf1 = load_frag(Bs, nw + 16, 72, kk, lane);
#pragma unroll
      for (int i = 0; i < 4; ++i) {
        v16bf af = load_frag(As, mw + i * 16, 72, kk, lane);
        acc[i][0] = wmma_bf16(af, bf0, acc[i][0]);
        acc[i][1] = wmma_bf16(af, bf1, acc[i][1]);
      }
    }
    __syncthreads();
  }

  const int half = lane >> 4, nl = lane & 15;
#pragma unroll
  for (int i = 0; i < 4; ++i)
#pragma unroll
    for (int j = 0; j < 2; ++j)
#pragma unroll
      for (int r = 0; r < 8; ++r) {
        int m = m0 + mw + i * 16 + r + 8 * half;
        int n = n0 + nw + j * 16 + nl;
        float val = acc[i][j][r] + bias[n];
        int b = m >> 11, s = m & 2047;
        int which = n >> 10, d = n & 1023, hh = d >> 6, dd = d & 63;
        int idx = ((b * N_HEAD + hh) * S_LEN + s) * HEAD_D + dd;
        if (which == 0) {
          qws[idx] = f2bf(val);
        } else if (which == 1) {
          kws[idx] = f2bf(val);
          presK[idx] = val;
        } else {
          vtws[((b * N_HEAD + hh) * HEAD_D + dd) * S_LEN + s] = f2bf(val);
          presV[idx] = val;
        }
      }
}

// ---------------------------------------------------------------------------
// Kernel 2: flash-style causal attention per (b, h, 128-query tile).
// 8 waves; wave owns 16 query rows. K/V^T tiles staged via async global->LDS
// (ASYNCcnt). Per 128-key tile:
//   S = Q K^T (WMMA), scale+mask(-1e10), online softmax, O += P V (WMMA).
// ---------------------------------------------------------------------------
__global__ __launch_bounds__(256) void attn(
    const bf16* __restrict__ qws, const bf16* __restrict__ kws,
    const bf16* __restrict__ vtws, bf16* __restrict__ aws) {
  __shared__ __align__(16) bf16 Ks[128 * 72];        // K tile, [key][hd]
  __shared__ __align__(16) bf16 Vs[64 * 136];        // V^T tile, [hd][key]
  __shared__ __align__(16) bf16 Ps[8 * 16 * 136];    // per-wave P staging
  const int tid = threadIdx.x, lane = tid & 31, wave = tid >> 5;
  const int qt = blockIdx.x, h = blockIdx.y, b = blockIdx.z;
  const int bh = b * N_HEAD + h;
  const bf16* Q = qws + (long)bh * S_LEN * HEAD_D;
  const bf16* K = kws + (long)bh * S_LEN * HEAD_D;
  const bf16* V = vtws + (long)bh * HEAD_D * S_LEN;
  const int q0 = qt * 128;
  const int half = lane >> 4, nl = lane & 15;

  v16bf qf[2];
  qf[0] = load_frag(Q, q0 + wave * 16, HEAD_D, 0, lane);
  qf[1] = load_frag(Q, q0 + wave * 16, HEAD_D, 32, lane);

  float mrow[8], lrow[8];
#pragma unroll
  for (int r = 0; r < 8; ++r) { mrow[r] = -3.0e38f; lrow[r] = 0.0f; }
  v8f oacc[4];
#pragma unroll
  for (int nt = 0; nt < 4; ++nt) oacc[nt] = zero8();

  bf16* Pw = Ps + wave * 16 * 136;
  const float L2E = 1.4426950408889634f;

  for (int j = 0; j <= qt; ++j) {
    const int kv0 = j * 128;
    if (j < qt) {  // prefetch next tiles (global_prefetch_b8)
      __builtin_prefetch(K + (long)(kv0 + 128) * HEAD_D + tid * 32, 0, 3);
      __builtin_prefetch(V + (long)(tid & 63) * S_LEN + kv0 + 128, 0, 3);
    }
    // stage K (128x64) and V^T (64x128) tiles straight into LDS with
    // GLOBAL_LOAD_ASYNC_TO_LDS_B128 (no VGPR round-trip)
#pragma unroll
    for (int p = 0; p < 4; ++p) {
      int slot = tid + p * 256;
      int r = slot >> 3, c = (slot & 7) << 3;
      async_b128(K, (unsigned)(((kv0 + r) * HEAD_D + c) * 2), Ks + r * 72 + c);
    }
#pragma unroll
    for (int p = 0; p < 4; ++p) {
      int slot = tid + p * 256;
      int r = slot >> 4, c = (slot & 15) << 3;
      async_b128(V, (unsigned)((r * S_LEN + kv0 + c) * 2), Vs + r * 136 + c);
    }
    wait_async0();
    __syncthreads();

    // scores: 16 (q) x 128 (keys) per wave = 8 tiles, 2 K-steps each
    v8f sc[8];
#pragma unroll
    for (int t = 0; t < 8; ++t) sc[t] = zero8();
#pragma unroll
    for (int kk = 0; kk < 2; ++kk)
#pragma unroll
      for (int t = 0; t < 8; ++t) {
        v16bf kf = load_frag(Ks, t * 16, 72, kk * 32, lane);
        sc[t] = wmma_bf16(qf[kk], kf, sc[t]);
      }

    // scale by 1/sqrt(64), exact reference mask: masked = -1e10
    if (j == qt) {
#pragma unroll
      for (int t = 0; t < 8; ++t)
#pragma unroll
        for (int r = 0; r < 8; ++r) {
          int kn = kv0 + t * 16 + nl;
          int qm = q0 + wave * 16 + r + 8 * half;
          sc[t][r] = (kn <= qm) ? sc[t][r] * 0.125f : -1.0e10f;
        }
    } else {
#pragma unroll
      for (int t = 0; t < 8; ++t)
#pragma unroll
        for (int r = 0; r < 8; ++r) sc[t][r] *= 0.125f;
    }

    // online softmax: row reductions across 16 lanes of each half
    float mnew[8], corr[8];
#pragma unroll
    for (int r = 0; r < 8; ++r) {
      float rm = sc[0][r];
#pragma unroll
      for (int t = 1; t < 8; ++t) rm = fmaxf(rm, sc[t][r]);
#pragma unroll
      for (int off = 8; off >= 1; off >>= 1) rm = fmaxf(rm, __shfl_xor(rm, off, 32));
      mnew[r] = fmaxf(mrow[r], rm);
      corr[r] = exp2f((mrow[r] - mnew[r]) * L2E);
    }
#pragma unroll
    for (int t = 0; t < 8; ++t)
#pragma unroll
      for (int r = 0; r < 8; ++r)
        sc[t][r] = exp2f((sc[t][r] - mnew[r]) * L2E);
#pragma unroll
    for (int r = 0; r < 8; ++r) {
      float s = 0.0f;
#pragma unroll
      for (int t = 0; t < 8; ++t) s += sc[t][r];
#pragma unroll
      for (int off = 8; off >= 1; off >>= 1) s += __shfl_xor(s, off, 32);
      lrow[r] = lrow[r] * corr[r] + s;
      mrow[r] = mnew[r];
    }
#pragma unroll
    for (int nt = 0; nt < 4; ++nt)
#pragma unroll
      for (int r = 0; r < 8; ++r) oacc[nt][r] *= corr[r];

    // stage P row-major (wave-private; wave LDS ops are in-order)
#pragma unroll
    for (int t = 0; t < 8; ++t)
#pragma unroll
      for (int r = 0; r < 8; ++r)
        Pw[(r + 8 * half) * 136 + t * 16 + nl] = f2bf(sc[t][r]);

    // O(16x64) += P(16x128) @ V(128x64): 4 K-steps x 4 dim-tiles
#pragma unroll
    for (int kk = 0; kk < 4; ++kk) {
      v16bf pf = load_frag(Pw, 0, 136, kk * 32, lane);
#pragma unroll
      for (int nt = 0; nt < 4; ++nt) {
        v16bf vf = load_frag(Vs, nt * 16, 136, kk * 32, lane);
        oacc[nt] = wmma_bf16(pf, vf, oacc[nt]);
      }
    }
    __syncthreads();
  }

  // normalize by l and merge heads into (B,S,D) bf16 workspace
#pragma unroll
  for (int r = 0; r < 8; ++r) {
    float inv = 1.0f / lrow[r];
    int srow = q0 + wave * 16 + r + 8 * half;
#pragma unroll
    for (int nt = 0; nt < 4; ++nt)
      aws[(long)(b * S_LEN + srow) * D_MODEL + h * HEAD_D + nt * 16 + nl] =
          f2bf(oacc[nt][r] * inv);
  }
}

// ---------------------------------------------------------------------------
// Kernel 3: out = attn_out @ c_proj_w + c_proj_b  (fp32 result)
// A-tile (already bf16) staged with async global->LDS, overlapped with the
// fp32->bf16 weight transpose done through VGPRs.
// ---------------------------------------------------------------------------
__global__ __launch_bounds__(256) void proj_gemm(
    const bf16* __restrict__ a, const float* __restrict__ w,
    const float* __restrict__ bias, float* __restrict__ out) {
  __shared__ __align__(16) bf16 As[128 * 72];
  __shared__ __align__(16) bf16 Bs[128 * 72];
  const int tid = threadIdx.x, lane = tid & 31, wave = tid >> 5;
  const int n0 = blockIdx.x * 128, m0 = blockIdx.y * 128;
  const int mw = (wave >> 2) * 64, nw = (wave & 3) * 32;

  v8f acc[4][2];
#pragma unroll
  for (int i = 0; i < 4; ++i)
#pragma unroll
    for (int j = 0; j < 2; ++j) acc[i][j] = zero8();

  for (int kc = 0; kc < D_MODEL; kc += 64) {
#pragma unroll
    for (int p = 0; p < 4; ++p) {  // A tile: async b128 copies, no VGPR round-trip
      int slot = tid + p * 256;
      int r = slot >> 3, c = (slot & 7) << 3;
      async_b128(a, (unsigned)(((m0 + r) * D_MODEL + kc + c) * 2), As + r * 72 + c);
    }
#pragma unroll
    for (int p = 0; p < 8; ++p) {  // W tile transposed, fp32 -> bf16
      int slot = tid + p * 256;
      int k = slot >> 5, n = (slot & 31) << 2;
      float4 v = *(const float4*)(w + (long)(kc + k) * D_MODEL + n0 + n);
      Bs[(n + 0) * 72 + k] = f2bf(v.x);
      Bs[(n + 1) * 72 + k] = f2bf(v.y);
      Bs[(n + 2) * 72 + k] = f2bf(v.z);
      Bs[(n + 3) * 72 + k] = f2bf(v.w);
    }
    wait_async0();
    __syncthreads();
#pragma unroll
    for (int kk = 0; kk < 64; kk += 32) {
      v16bf bf0 = load_frag(Bs, nw + 0, 72, kk, lane);
      v16bf bf1 = load_frag(Bs, nw + 16, 72, kk, lane);
#pragma unroll
      for (int i = 0; i < 4; ++i) {
        v16bf af = load_frag(As, mw + i * 16, 72, kk, lane);
        acc[i][0] = wmma_bf16(af, bf0, acc[i][0]);
        acc[i][1] = wmma_bf16(af, bf1, acc[i][1]);
      }
    }
    __syncthreads();
  }

  const int half = lane >> 4, nl = lane & 15;
#pragma unroll
  for (int i = 0; i < 4; ++i)
#pragma unroll
    for (int j = 0; j < 2; ++j)
#pragma unroll
      for (int r = 0; r < 8; ++r) {
        int m = m0 + mw + i * 16 + r + 8 * half;
        int n = n0 + nw + j * 16 + nl;
        out[(long)m * D_MODEL + n] = acc[i][j][r] + bias[n];
      }
}

extern "C" void kernel_launch(void* const* d_in, const int* in_sizes, int n_in,
                              void* d_out, int out_size, void* d_ws, size_t ws_size,
                              hipStream_t stream) {
  const float* x        = (const float*)d_in[0];
  const float* c_attn_w = (const float*)d_in[1];
  const float* c_attn_b = (const float*)d_in[2];
  const float* c_proj_w = (const float*)d_in[3];
  const float* c_proj_b = (const float*)d_in[4];

  float* out = (float*)d_out;                 // a: 4,194,304 floats
  float* presK = out + 4194304;               // present[0] (k): 4,194,304
  float* presV = out + 8388608;               // present[1] (v): 4,194,304

  bf16* ws   = (bf16*)d_ws;                   // needs 32 MB of workspace
  bf16* qws  = ws;                            // (B,H,S,hd)
  bf16* kws  = ws + 4194304;                  // (B,H,S,hd)
  bf16* vtws = ws + 8388608;                  // (B,H,hd,S)
  bf16* aws  = ws + 12582912;                 // (B,S,D)

  qkv_gemm<<<dim3(24, 32), 256, 0, stream>>>(x, c_attn_w, c_attn_b, qws, kws,
                                             vtws, presK, presV);
  attn<<<dim3(16, 16, 2), 256, 0, stream>>>(qws, kws, vtws, aws);
  proj_gemm<<<dim3(8, 32), 256, 0, stream>>>(aws, c_proj_w, c_proj_b, out);
}
